// NODE_RNN_53283364274894
// MI455X (gfx1250) — matrix operand order repaired and checked by reading, not compiled, not run on hardware
//
#include <hip/hip_runtime.h>
#include <math.h>

typedef __attribute__((ext_vector_type(16))) _Float16 v16h;
typedef __attribute__((ext_vector_type(8)))  float    v8f;

namespace {
constexpr int kB    = 256;   // batch
constexpr int kT    = 512;   // timesteps (only last one is live)
constexpr int kDin  = 64;
constexpr int kHid  = 64;
constexpr int kNb   = 10;    // real basis count
constexpr int kNbp  = 16;    // padded basis count -> K slots per feature
constexpr int kCls  = 2;
constexpr int kRows = 16;    // M-tile rows per block
}

// Hardware tanh (v_tanh_f32 is a CDNA5 TRANS op; co-executes with WMMA).
__device__ __forceinline__ float fast_tanh(float v) {
#if __has_builtin(__builtin_amdgcn_tanhf)
    return __builtin_amdgcn_tanhf(v);
#else
    // Branch-free fallback on v_exp_f32
    const float t = __expf(-2.0f * __builtin_fabsf(v));
    const float y = (1.0f - t) / (1.0f + t);
    return __builtin_copysignf(y, v);
#endif
}

__device__ __forceinline__ float fast_sigmoid(float v) {
    return 1.0f / (1.0f + __expf(-v));   // v_exp_f32, branch-free
}

// One ferro-basis layer for a 16-row tile:
//   out[m, n] = sum_{i<64, kb<10} tanh((in[m,i]-ctr[kb])/w[kb]) * coeffs[i,kb,n]
// K padded to 64*16=1024, consumed as 32 chunks of K=32 (2 features/chunk)
// via v_wmma_f32_16x16x32_f16. Each of the 4 waves owns one 16-col N-tile.
template <bool SIGMOID>
__device__ __forceinline__ void kan_stage(const float* __restrict__ centers,
                                          const float* __restrict__ widths,
                                          const float* __restrict__ coeffs, // (64, 10, 64) f32
                                          const float* sin_buf,             // LDS in  (16 x 64)
                                          float*       sout_buf,            // LDS out (16 x 64)
                                          int lane, int wave)
{
    const int half = lane >> 4;        // which 16-lane half of the wave
    const int mrow = lane & 15;        // A-fragment row for this lane
    const int ncol = wave * 16 + mrow; // B/D column for this lane

    float ctr[8], invw[8], kmask[8];
#pragma unroll
    for (int j = 0; j < 8; ++j) {
        const int kb = half * 8 + j;           // this lane's basis slots
        const bool live = (kb < kNb);
        ctr[j]   = live ? centers[kb]         : 0.0f;
        invw[j]  = live ? (1.0f / widths[kb]) : 0.0f;
        kmask[j] = live ? 1.0f : 0.0f;
    }

    v8f acc = {};
#pragma unroll 4
    for (int c2 = 0; c2 < kHid / 2; ++c2) {
        // ---- A fragment: 16x32 f16, ISA layout ----
        // lane half h: elem j<8 -> K=8h+j (feat 2c), elem j+8 -> K=16+8h+j (feat 2c+1)
        const float x0 = sin_buf[mrow * kHid + 2 * c2 + 0];
        const float x1 = sin_buf[mrow * kHid + 2 * c2 + 1];
        v16h a;
#pragma unroll
        for (int j = 0; j < 8; ++j) {
            const float p0 = kmask[j] * fast_tanh((x0 - ctr[j]) * invw[j]);
            const float p1 = kmask[j] * fast_tanh((x1 - ctr[j]) * invw[j]);
            a[j]     = (_Float16)p0;
            a[j + 8] = (_Float16)p1;
        }

        // ---- B fragment: 32x16 f16, ISA layout ----
        // lane half h: elem j<8 -> feature 2c, kb = 8h+j; elem j+8 -> feature 2c+1, kb = 8h+j; col = ncol
        const int    feat = 2 * c2;
        const float* kp   = coeffs + ((size_t)feat * kNb) * kHid + ncol;
        v16h bf;
#pragma unroll
        for (int j = 0; j < kNbp; ++j) {
            const int kbj = half * 8 + (j & 7);
            const float v = (kbj < kNb) ? kp[((size_t)(j >> 3) * kNb + kbj) * kHid] : 0.0f;
            bf[j] = (_Float16)v;
        }

        acc = __builtin_amdgcn_wmma_f32_16x16x32_f16(
            /*neg_a=*/false, a, /*neg_b=*/false, bf,
            /*c_mod=*/(short)0, acc, /*reuse_a=*/false, /*reuse_b=*/false);
    }

    // D layout: VGPR r, lane half h -> row (h ? r+8 : r), col = lane&15 within N-tile
#pragma unroll
    for (int r = 0; r < 8; ++r) {
        float v = acc[r];
        if (SIGMOID) v = fast_sigmoid(v);
        const int m = half ? (r + 8) : r;
        sout_buf[m * kHid + ncol] = v;
    }
    __syncthreads();
}

__global__ __launch_bounds__(128, 1) void kan_node_fused(
    const float* __restrict__ x,
    const float* __restrict__ lift_c, const float* __restrict__ lift_w, const float* __restrict__ lift_k,
    const float* __restrict__ rnn_c,  const float* __restrict__ rnn_w,  const float* __restrict__ rnn_k,
    const float* __restrict__ head_c, const float* __restrict__ head_w, const float* __restrict__ head_k,
    const float* __restrict__ headW,  const float* __restrict__ headb,
    float* __restrict__ out)
{
    __shared__ float sbuf0[kRows * kHid];
    __shared__ float sbuf1[kRows * kHid];

    const int tid  = threadIdx.x;
    const int wave = tid >> 5;   // N-tile owner (0..3)
    const int lane = tid & 31;
    const int m0   = blockIdx.x * kRows;

    // Stage in x[:, T-1, :] tile (the only live timestep), coalesced.
    for (int idx = tid; idx < kRows * kDin; idx += 128) {
        const int r = idx >> 6;
        const int c = idx & 63;
        sbuf0[idx] = x[((size_t)(m0 + r) * kT + (kT - 1)) * kDin + c];
    }
    __syncthreads();

    kan_stage<false>(lift_c, lift_w, lift_k, sbuf0, sbuf1, lane, wave); // z   = lift(x_last)
    kan_stage<true >(rnn_c,  rnn_w,  rnn_k,  sbuf1, sbuf0, lane, wave); // h   = sigmoid(rnn_basis(z))
    kan_stage<true >(head_c, head_w, head_k, sbuf0, sbuf1, lane, wave); // phi = sigmoid(head_basis(h))

    // logits = phi @ head_W + head_b  (16x64 @ 64x2, exact f32)
    if (tid < kRows * kCls) {
        const int r = tid >> 1;
        const int c = tid & 1;
        float accv = headb[c];
#pragma unroll 8
        for (int i = 0; i < kHid; ++i)
            accv = fmaf(sbuf1[r * kHid + i], headW[i * kCls + c], accv);
        out[(size_t)(m0 + r) * kCls + c] = accv;
    }
}

extern "C" void kernel_launch(void* const* d_in, const int* in_sizes, int n_in,
                              void* d_out, int out_size, void* d_ws, size_t ws_size,
                              hipStream_t stream) {
    (void)in_sizes; (void)n_in; (void)out_size; (void)d_ws; (void)ws_size;
    const float* x      = (const float*)d_in[0];
    const float* lift_c = (const float*)d_in[1];
    const float* lift_w = (const float*)d_in[2];
    const float* lift_k = (const float*)d_in[3];
    const float* rnn_c  = (const float*)d_in[4];
    const float* rnn_w  = (const float*)d_in[5];
    const float* rnn_k  = (const float*)d_in[6];
    const float* head_c = (const float*)d_in[7];
    const float* head_w = (const float*)d_in[8];
    const float* head_k = (const float*)d_in[9];
    const float* headW  = (const float*)d_in[10];
    const float* headb  = (const float*)d_in[11];
    float* outp = (float*)d_out;

    dim3 grid(kB / kRows);   // 16 blocks
    dim3 block(128);         // 4 waves (wave32)
    hipLaunchKernelGGL(kan_node_fused, grid, block, 0, stream,
                       x, lift_c, lift_w, lift_k,
                       rnn_c, rnn_w, rnn_k,
                       head_c, head_w, head_k,
                       headW, headb, outp);
}
